// HLOCNetVLAD_38371237823140
// MI455X (gfx1250) — matrix operand order repaired and checked by reading, not compiled, not run on hardware
//
#include <hip/hip_runtime.h>
#include <hip/hip_bf16.h>

// ---------------------------------------------------------------------------
// NetVLAD fused kernel for MI455X (gfx1250, wave32, WMMA f16 16x16x32).
// B=32, D=512, K=64, N=1024.  Memory-bound (~60 FLOP/B) -> fuse both GEMMs
// through LDS so x (64 MB fp32) is read from HBM exactly once.
// ---------------------------------------------------------------------------

typedef __attribute__((ext_vector_type(16))) _Float16 v16h;
typedef __attribute__((ext_vector_type(8)))  float    v8f;

union U16h { v16h v; _Float16 h[16]; uint4 q[2]; };
union U8f  { v8f  v; float    f[8]; };

#define BB    32
#define DD    512
#define KK    64
#define NNN   1024
#define NT    32      // N columns per chunk
#define CH    2       // chunks per workgroup (NT*CH = 64 cols per WG)
#define SPLIT 16      // N-splits per batch (16*64 = 1024)
#define PADH  8       // LDS row pad (halves) to break bank conflicts

// ---------------- kernel 1: conv_w fp32 -> fp16 -----------------------------
__global__ __launch_bounds__(256) void netvlad_convw(
    const float* __restrict__ w, _Float16* __restrict__ wh, int n)
{
    int i = blockIdx.x * 256 + threadIdx.x;
    if (i < n) wh[i] = (_Float16)w[i];
}

// ---------------- kernel 2: fused scores -> softmax -> vlad -----------------
__global__ __launch_bounds__(256) void netvlad_fused(
    const float* __restrict__ x,      // [B, D, N] fp32
    const _Float16* __restrict__ wh,  // [K, D] fp16 row-major
    float* __restrict__ vlad,         // [B, D, K] fp32 accumulators (zeroed)
    float* __restrict__ asum)         // [B, K] fp32 accumulators (zeroed)
{
    __shared__ alignas(32) _Float16 lx[NT][DD + PADH];  // x tile, d-contig
    __shared__ float                lsc[KK][NT + 1];    // score tile
    __shared__ alignas(32) _Float16 las[KK][NT];        // assign tile, n-contig
    __shared__ float                lred[8][NT];
    __shared__ float                lrow[NT];

    const int b     = blockIdx.y;
    const int split = blockIdx.x;
    const int t     = (int)threadIdx.x;
    const int lane  = t & 31;
    const int wid   = t >> 5;           // 8 waves
    const float* xb = x + (size_t)b * DD * NNN;

    // vlad register accumulators: wave wid owns d-tiles [4*wid .. 4*wid+3],
    // all 4 k-tiles.  4*4 tiles * v8f = 128 VGPRs/lane.
    U8f accV[4][4];
#pragma unroll
    for (int i = 0; i < 4; ++i)
#pragma unroll
        for (int j = 0; j < 4; ++j)
#pragma unroll
            for (int e = 0; e < 8; ++e) accV[i][j].f[e] = 0.0f;
    float asumReg = 0.0f;

    for (int ch = 0; ch < CH; ++ch) {
        const int n0 = split * (NT * CH) + ch * NT;

        // ---- stage x chunk (fp32 HBM -> fp16 LDS), coalesced 128B rows ----
        for (int r = wid; r < DD; r += 8) {
            lx[lane][r] = (_Float16)xb[(size_t)r * NNN + n0 + lane];
            if (ch == 0)  // prefetch next chunk's cache lines
                __builtin_prefetch(xb + (size_t)r * NNN + n0 + NT + lane, 0, 1);
        }
        __syncthreads();

        // ---- phase 2: scores(64 x NT) = W(64 x D) * xchunk(D x NT) --------
        {
            const int m    = wid >> 1;                 // k-cluster tile 0..3
            const int ns   = wid & 1;                  // n sub-tile 0..1
            const int row  = m * 16 + (lane & 15);     // A row (cluster)
            const int kb   = (lane >> 4) << 3;         // A K-offset 0/8
            const int nloc = ns * 16 + (lane & 15);    // B column (n)
            const int db   = (lane >> 4) << 3;         // B K-offset 0/8
            U8f accS;
#pragma unroll
            for (int e = 0; e < 8; ++e) accS.f[e] = 0.0f;
#pragma unroll 4
            for (int c = 0; c < DD / 32; ++c) {
                U16h A, Bf;
                // A fragment: 16-bit 16x32 layout (K 0-7 & 16-23 / 8-15 & 24-31)
                const uint4* wp = (const uint4*)(wh + row * DD + c * 32 + kb);
                A.q[0] = wp[0];
                A.q[1] = wp[2];
                // B fragment: column n, same K layout as A (8 halves at db, 8 at db+16)
                const uint4* bp = (const uint4*)(&lx[nloc][c * 32 + db]);
                Bf.q[0] = bp[0];
                Bf.q[1] = bp[2];
                accS.v = __builtin_amdgcn_wmma_f32_16x16x32_f16(
                    false, A.v, false, Bf.v, (short)0, accS.v, false, false);
            }
            // C layout: lanes 0-15 -> M=j, lanes 16-31 -> M=8+j
#pragma unroll
            for (int j = 0; j < 8; ++j)
                lsc[m * 16 + ((lane >> 4) << 3) + j][ns * 16 + (lane & 15)] =
                    accS.f[j];
        }
        __syncthreads();

        // ---- phase 3: softmax over K=64 per column ------------------------
        {
            const int col = t & 31;
            const int qg  = t >> 5;   // 8 groups of 8 clusters
            float mx = lsc[qg * 8 + 0][col];
#pragma unroll
            for (int j = 1; j < 8; ++j) mx = fmaxf(mx, lsc[qg * 8 + j][col]);
            lred[qg][col] = mx;
            __syncthreads();
            if (t < NT) {
                float m2 = lred[0][t];
#pragma unroll
                for (int j = 1; j < 8; ++j) m2 = fmaxf(m2, lred[j][t]);
                lrow[t] = m2;
            }
            __syncthreads();
            const float bm = lrow[col];
            float s = 0.0f;
#pragma unroll
            for (int j = 0; j < 8; ++j) s += __expf(lsc[qg * 8 + j][col] - bm);
            lred[qg][col] = s;
            __syncthreads();
            if (t < NT) {
                float s2 = 0.0f;
#pragma unroll
                for (int j = 0; j < 8; ++j) s2 += lred[j][t];
                lrow[t] = 1.0f / s2;
            }
            __syncthreads();
            const float inv = lrow[col];
#pragma unroll
            for (int j = 0; j < 8; ++j)
                las[qg * 8 + j][col] =
                    (_Float16)(__expf(lsc[qg * 8 + j][col] - bm) * inv);
        }
        __syncthreads();

        // ---- per-cluster assign sums (for centers term) -------------------
        if (t < KK) {
            float ss = 0.0f;
            for (int n = 0; n < NT; ++n) ss += (float)las[t][n];
            asumReg += ss;
        }

        // ---- phase 4: vlad(D x K) += xchunk(D x NT) * assign^T(NT x K) ----
#pragma unroll
        for (int i = 0; i < 4; ++i) {
            const int dt = wid * 4 + i;
            U16h A;
            const int d  = dt * 16 + (lane & 15);   // A row (descriptor dim)
            const int nb = (lane >> 4) << 3;        // A K(n)-offset 0/8
#pragma unroll
            for (int j = 0; j < 8; ++j) A.h[j]     = lx[nb + j][d];
#pragma unroll
            for (int j = 0; j < 8; ++j) A.h[8 + j] = lx[nb + 16 + j][d];
#pragma unroll
            for (int kt = 0; kt < 4; ++kt) {
                U16h Bf;
                const int kcol = kt * 16 + (lane & 15); // B column (cluster)
                const int nb2  = (lane >> 4) << 3;      // B K(n)-offset 0/8
                const uint4* bp = (const uint4*)(&las[kcol][nb2]);
                Bf.q[0] = bp[0];
                Bf.q[1] = bp[2];
                accV[i][kt].v = __builtin_amdgcn_wmma_f32_16x16x32_f16(
                    false, A.v, false, Bf.v, (short)0, accV[i][kt].v,
                    false, false);
            }
        }
        __syncthreads();   // protect lx/las before next chunk overwrites
    }

    // ---- flush register accumulators with f32 atomics (L2-resident) -------
    float* vb = vlad + (size_t)b * DD * KK;
#pragma unroll
    for (int i = 0; i < 4; ++i) {
        const int dt = wid * 4 + i;
#pragma unroll
        for (int kt = 0; kt < 4; ++kt) {
            const int k = kt * 16 + (lane & 15);
#pragma unroll
            for (int j = 0; j < 8; ++j) {
                const int d = dt * 16 + ((lane >> 4) << 3) + j;
                atomicAdd(&vb[d * KK + k], accV[i][kt].f[j]);
            }
        }
    }
    if (t < KK) atomicAdd(&asum[b * KK + t], asumReg);
}

// ---------------- kernel 3: centers term + dual L2 normalize ----------------
__global__ __launch_bounds__(256) void netvlad_finalize(
    const float* __restrict__ vlad, const float* __restrict__ asum,
    const float* __restrict__ centers, float* __restrict__ out)
{
    __shared__ float lasum[KK];
    __shared__ float lsq[4][KK];
    __shared__ float linv[KK];
    __shared__ float lgi;
    const int b = blockIdx.x;
    const int t = (int)threadIdx.x;
    const float* vb = vlad + (size_t)b * DD * KK;
    if (t < KK) lasum[t] = asum[b * KK + t];
    __syncthreads();

    // per-cluster sum of squares of (vlad - centers*asum)
    const int k    = t & 63;
    const int part = t >> 6;    // 4 partials per cluster
    const float a  = lasum[k];
    float s = 0.0f;
    for (int d = part * 128; d < part * 128 + 128; ++d) {
        const float v = vb[d * KK + k] - centers[d * KK + k] * a;
        s += v * v;
    }
    lsq[part][k] = s;
    __syncthreads();
    if (t < KK) {
        const float tot = lsq[0][t] + lsq[1][t] + lsq[2][t] + lsq[3][t];
        const float iv  = 1.0f / fmaxf(sqrtf(tot), 1e-12f);
        linv[t]   = iv;
        lsq[0][t] = tot * iv * iv;   // this cluster's contribution post-norm
    }
    __syncthreads();
    if (t == 0) {
        float g = 0.0f;
        for (int kk = 0; kk < KK; ++kk) g += lsq[0][kk];
        lgi = 1.0f / fmaxf(sqrtf(g), 1e-12f);
    }
    __syncthreads();
    const float gi = lgi;
    for (int e = t; e < DD * KK; e += 256) {
        const int k2  = e & 63;
        const float v = vb[e] - centers[e] * lasum[k2];
        out[(size_t)b * DD * KK + e] = v * linv[k2] * gi;
    }
}

// ---------------------------------------------------------------------------
extern "C" void kernel_launch(void* const* d_in, const int* in_sizes, int n_in,
                              void* d_out, int out_size, void* d_ws,
                              size_t ws_size, hipStream_t stream)
{
    (void)in_sizes; (void)n_in; (void)out_size; (void)ws_size;
    const float* x       = (const float*)d_in[0];  // [B, D, N]
    const float* conv_w  = (const float*)d_in[1];  // [K, D]
    const float* centers = (const float*)d_in[2];  // [D, K]
    float* out = (float*)d_out;                    // [B, D*K]

    // workspace layout: vlad f32 [B*D*K] | asum f32 [B*K] | wh f16 [K*D]
    float*    vlad = (float*)d_ws;
    float*    asum = vlad + (size_t)BB * DD * KK;
    _Float16* wh   = (_Float16*)(asum + BB * KK);

    // accumulators must be re-zeroed every call (graph replays)
    hipMemsetAsync(d_ws, 0, ((size_t)BB * DD * KK + BB * KK) * sizeof(float),
                   stream);

    netvlad_convw<<<(KK * DD + 255) / 256, 256, 0, stream>>>(conv_w, wh,
                                                             KK * DD);

    dim3 g2(SPLIT, BB);
    netvlad_fused<<<g2, 256, 0, stream>>>(x, wh, vlad, asum);

    netvlad_finalize<<<BB, 256, 0, stream>>>(vlad, asum, centers, out);
}
